// Self_Attention_6768868458812
// MI455X (gfx1250) — matrix operand order repaired and checked by reading, not compiled, not run on hardware
//
#include <hip/hip_runtime.h>
#include <hip/hip_bf16.h>

// Problem constants (match the reference: B=32, L=2048, D=1024).
#define BB 32
#define LL 2048
#define DD 1024
#define CHUNK 128           // rows of L handled per pass-3 block
#define NCH (LL / CHUNK)    // 16 partial chunks per batch

typedef __attribute__((ext_vector_type(16))) _Float16 v16h;
typedef __attribute__((ext_vector_type(8)))  float    v8f;

// ---------------------------------------------------------------------------
// Pass 1: f1[b,l] = dot(E[b,l,:], w1); f2[b,l] = dot(E[b,l,:], w2)
// One wave per 16-row tile. WMMA: C[16,16] += A[16x32] x B[32x16] where
//   A = f16(E rows l0..l0+15, cols kk..kk+31)
//   B column 0 = f16(w1[kk..kk+31]), column 1 = f16(w2[...]), rest 0.
// C col 0 accumulates f1, col 1 accumulates f2 (f32 accumulate).
// A layout (16-bit A 16x32, ISA 7.12.2): lanes 0-15 = rows M, K={0..7,16..23};
// lanes 16-31 = rows M, K={8..15,24..31}.  C layout: VGPR r -> row r (lanes
// 0-15) / row r+8 (lanes 16-31), N = lane&15.
// Each iteration consumes a full, aligned 128B line per row (cols kk..kk+31),
// so every fetched byte is used: pure streaming at HBM rate.
// ---------------------------------------------------------------------------
__global__ __launch_bounds__(256) void gemv_f12_wmma(
    const float* __restrict__ emb, const float* __restrict__ w1,
    const float* __restrict__ w2, float* __restrict__ f1,
    float* __restrict__ f2) {
  const int lane = threadIdx.x & 31;
  const int waveInBlk = threadIdx.x >> 5;
  const int tile = blockIdx.x * 8 + waveInBlk;  // 0 .. B*L/16-1 (=4096 waves)
  const int b  = tile >> 7;                     // tile / (L/16)
  const int l0 = (tile & 127) << 4;
  const int m  = lane & 15;
  const int hi = lane >> 4;

  const float* Erow = emb + ((size_t)b * LL + (size_t)(l0 + m)) * DD;

  v8f acc = {0.f, 0.f, 0.f, 0.f, 0.f, 0.f, 0.f, 0.f};

  // B operand: zero once outside the loop.  Lanes m<2 overwrite all 16
  // elements every iteration; lanes m>=2 keep the loop-invariant zeros.
  v16h Bv;
#pragma unroll
  for (int j = 0; j < 16; ++j) Bv[j] = (_Float16)0.0f;

  for (int kk = 0; kk < DD; kk += 32) {
    // ---- A tile: 16 f32 per lane -> v16h (K groups per 7.12.2) ----
    const float4 a0 = *(const float4*)(Erow + kk + hi * 8);
    const float4 a1 = *(const float4*)(Erow + kk + hi * 8 + 4);
    const float4 a2 = *(const float4*)(Erow + kk + hi * 8 + 16);
    const float4 a3 = *(const float4*)(Erow + kk + hi * 8 + 20);
    v16h A;
    A[0]  = (_Float16)a0.x; A[1]  = (_Float16)a0.y;
    A[2]  = (_Float16)a0.z; A[3]  = (_Float16)a0.w;
    A[4]  = (_Float16)a1.x; A[5]  = (_Float16)a1.y;
    A[6]  = (_Float16)a1.z; A[7]  = (_Float16)a1.w;
    A[8]  = (_Float16)a2.x; A[9]  = (_Float16)a2.y;
    A[10] = (_Float16)a2.z; A[11] = (_Float16)a2.w;
    A[12] = (_Float16)a3.x; A[13] = (_Float16)a3.y;
    A[14] = (_Float16)a3.z; A[15] = (_Float16)a3.w;

    // ---- B tile: column 0 = w1, column 1 = w2 (cache-resident, 4KB each) ---
    if (m < 2) {
      const float* w = (m == 0) ? w1 : w2;
      const float4 b0 = *(const float4*)(w + kk + hi * 8);
      const float4 b1 = *(const float4*)(w + kk + hi * 8 + 4);
      const float4 b2 = *(const float4*)(w + kk + hi * 8 + 16);
      const float4 b3 = *(const float4*)(w + kk + hi * 8 + 20);
      Bv[0]  = (_Float16)b0.x; Bv[1]  = (_Float16)b0.y;
      Bv[2]  = (_Float16)b0.z; Bv[3]  = (_Float16)b0.w;
      Bv[4]  = (_Float16)b1.x; Bv[5]  = (_Float16)b1.y;
      Bv[6]  = (_Float16)b1.z; Bv[7]  = (_Float16)b1.w;
      Bv[8]  = (_Float16)b2.x; Bv[9]  = (_Float16)b2.y;
      Bv[10] = (_Float16)b2.z; Bv[11] = (_Float16)b2.w;
      Bv[12] = (_Float16)b3.x; Bv[13] = (_Float16)b3.y;
      Bv[14] = (_Float16)b3.z; Bv[15] = (_Float16)b3.w;
    }
    // (neg_a, A, neg_b, B, c_mod, C, reuse_a, reuse_b)
    acc = __builtin_amdgcn_wmma_f32_16x16x32_f16(
        false, A, false, Bv, (short)0, acc, false, false);
  }

  // C extraction: lane (hi, n=m) holds C[hi*8 + r][m] in acc[r].
  if (m < 2) {
    float* dst = (m == 0) ? f1 : f2;
#pragma unroll
    for (int r = 0; r < 8; ++r)
      dst[(size_t)b * LL + l0 + hi * 8 + r] = acc[r];
  }
}

// ---------------------------------------------------------------------------
// Pass 2: per-batch top-2 of f2, logits, softmax -> a[b,l].  One block/batch.
// Tiny (32 x 2048); deterministic thread-0 reductions.
// ---------------------------------------------------------------------------
__global__ __launch_bounds__(256) void softmax_weights(
    const float* __restrict__ f1, const float* __restrict__ f2,
    const float* __restrict__ bias, float* __restrict__ aout) {
  __shared__ float sv1[256];
  __shared__ float sv2[256];
  __shared__ int   si1[256];
  __shared__ float sred[256];
  __shared__ float bc[4];
  __shared__ int   bgi;

  const int b = blockIdx.x, t = threadIdx.x;
  const float* F1 = f1 + (size_t)b * LL;
  const float* F2 = f2 + (size_t)b * LL;

  // local top-2 of f2 (strided over L)
  float v1 = -INFINITY, v2 = -INFINITY;
  int i1 = 0x7fffffff;
  for (int l = t; l < LL; l += 256) {
    float v = F2[l];
    if (v > v1) { v2 = v1; v1 = v; i1 = l; }
    else if (v > v2) { v2 = v; }
  }
  sv1[t] = v1; sv2[t] = v2; si1[t] = i1;
  __syncthreads();
  if (t == 0) {
    float g1 = -INFINITY, g2 = -INFINITY;
    int gi = 0x7fffffff;
    for (int i = 0; i < 256; ++i) {
      float c1 = sv1[i], c2 = sv2[i];
      int ci = si1[i];
      if (c1 > g1) {
        g2 = (g1 > c2) ? g1 : c2;
        g1 = c1; gi = ci;
      } else {
        if (c1 == g1 && ci < gi) gi = ci;  // match top_k first-occurrence on tie
        if (c1 > g2) g2 = c1;
        if (c2 > g2) g2 = c2;
      }
    }
    bc[0] = g1; bc[1] = g2; bgi = gi;
  }
  __syncthreads();
  const float g1 = bc[0], g2 = bc[1];
  const int gi = bgi;
  const float bs = bias[0];

  // row max of logits
  float lm = -INFINITY;
  for (int l = t; l < LL; l += 256) {
    float off = (l == gi) ? g2 : g1;
    lm = fmaxf(lm, F1[l] + off + bs);
  }
  sred[t] = lm;
  __syncthreads();
  if (t == 0) {
    float M = -INFINITY;
    for (int i = 0; i < 256; ++i) M = fmaxf(M, sred[i]);
    bc[2] = M;
  }
  __syncthreads();
  const float M = bc[2];

  // sum of exp
  float ls = 0.f;
  for (int l = t; l < LL; l += 256) {
    float off = (l == gi) ? g2 : g1;
    ls += __expf(F1[l] + off + bs - M);
  }
  sred[t] = ls;
  __syncthreads();
  if (t == 0) {
    float Z = 0.f;
    for (int i = 0; i < 256; ++i) Z += sred[i];
    bc[3] = Z;
  }
  __syncthreads();
  const float invZ = 1.0f / bc[3];

  float* A = aout + (size_t)b * LL;
  for (int l = t; l < LL; l += 256) {
    float off = (l == gi) ? g2 : g1;
    A[l] = __expf(F1[l] + off + bs - M) * invZ;
  }
}

// ---------------------------------------------------------------------------
// Pass 3: partial weighted sums  part[b,c,d] = sum_{l in chunk c} a[b,l]*E[b,l,d]
// Grid (B, NCH) = 512 blocks, 256 threads; thread t owns 4 consecutive d
// (float4 loads, full 128B-line streaming).  global_prefetch_b8 hides HBM
// latency.  Batches are processed in REVERSE order: pass 1 streamed batches
// 0->31, so L2 (192MB) still holds the tail batches when pass 3 starts.
// ---------------------------------------------------------------------------
__global__ __launch_bounds__(256) void weighted_sum_part(
    const float* __restrict__ emb, const float* __restrict__ a,
    float* __restrict__ part) {
  const int b = (BB - 1) - blockIdx.x;   // reverse for L2 reuse
  const int c = blockIdx.y;
  const int t = threadIdx.x;
  const int d = t * 4;
  const int l0 = c * CHUNK;

  const float* Ebase = emb + ((size_t)b * LL + l0) * DD + d;
  const float* Av = a + (size_t)b * LL + l0;

  float acx = 0.f, acy = 0.f, acz = 0.f, acw = 0.f;
  for (int l = 0; l < CHUNK; ++l) {
    if (l + 8 < CHUNK)
      __builtin_prefetch(Ebase + (size_t)(l + 8) * DD, 0, 3);
    const float w = Av[l];
    const float4 e = *(const float4*)(Ebase + (size_t)l * DD);
    acx += w * e.x; acy += w * e.y; acz += w * e.z; acw += w * e.w;
  }
  float4 out;
  out.x = acx; out.y = acy; out.z = acz; out.w = acw;
  *(float4*)(part + ((size_t)(b * NCH + c)) * DD + d) = out;
}

// ---------------------------------------------------------------------------
// Pass 4: deterministic reduction of the NCH partials -> out[b,d]
// ---------------------------------------------------------------------------
__global__ __launch_bounds__(256) void reduce_out(
    const float* __restrict__ part, float* __restrict__ out) {
  const int idx = blockIdx.x * 256 + threadIdx.x;  // 0 .. B*D-1
  const int b = idx >> 10;
  const int d = idx & (DD - 1);
  float s = 0.f;
#pragma unroll
  for (int c = 0; c < NCH; ++c)
    s += part[((size_t)(b * NCH + c) << 10) + d];
  out[idx] = s;
}

extern "C" void kernel_launch(void* const* d_in, const int* in_sizes, int n_in,
                              void* d_out, int out_size, void* d_ws,
                              size_t ws_size, hipStream_t stream) {
  const float* emb  = (const float*)d_in[0];  // [B,L,D] f32
  const float* w1   = (const float*)d_in[1];  // [D,1]   f32
  const float* w2   = (const float*)d_in[2];  // [D,1]   f32
  const float* bias = (const float*)d_in[3];  // [1]     f32
  float* out = (float*)d_out;                 // [B,D]   f32

  float* ws = (float*)d_ws;
  float* f1   = ws;                        // B*L
  float* f2   = ws + (size_t)BB * LL;      // B*L
  float* aw   = ws + (size_t)2 * BB * LL;  // B*L
  float* part = ws + (size_t)3 * BB * LL;  // B*NCH*D  (~2MB)

  // Pass 1: 4096 tile-waves, 8 waves/block.
  gemv_f12_wmma<<<(BB * (LL / 16)) / 8, 256, 0, stream>>>(emb, w1, w2, f1, f2);
  // Pass 2: one block per batch.
  softmax_weights<<<BB, 256, 0, stream>>>(f1, f2, bias, aw);
  // Pass 3: partial weighted sums (512 independent memory streams).
  dim3 g3(BB, NCH);
  weighted_sum_part<<<g3, 256, 0, stream>>>(emb, aw, part);
  // Pass 4: deterministic final reduction.
  reduce_out<<<(BB * DD) / 256, 256, 0, stream>>>(part, out);
}
